// GCN_9122510536818
// MI455X (gfx1250) — matrix operand order repaired and checked by reading, not compiled, not run on hardware
//
#include <hip/hip_runtime.h>
#include <hip/hip_bf16.h>

// ---------------- problem constants ----------------
#define N_NODES   50000
#define SEQ_LEN   32
#define EMB_DIM   128
#define IN_FEATS  256
#define N_HIDDEN  256
#define N_CLASSES 40
#define N_CLS_PAD 48
#define N_EDGES   800000

typedef __attribute__((ext_vector_type(16))) _Float16 v16h;
typedef __attribute__((ext_vector_type(8)))  float    v8f;

// ---------------- degree / norm kernels ----------------
__global__ void edge_degree_kernel(const int* __restrict__ src,
                                   const int* __restrict__ dst,
                                   float* __restrict__ deg_s,
                                   float* __restrict__ deg_d) {
    int e = blockIdx.x * blockDim.x + threadIdx.x;
    if (e < N_EDGES) {
        atomicAdd(&deg_s[src[e]], 1.0f);
        atomicAdd(&deg_d[dst[e]], 1.0f);
    }
}

__global__ void norm_kernel(float* __restrict__ deg, int n) {
    int i = blockIdx.x * blockDim.x + threadIdx.x;
    if (i < n) deg[i] = rsqrtf(fmaxf(deg[i], 1.0f));
}

// ---------------- embedding pool: one wave per node ----------------
__global__ void embed_pool_kernel(const int* __restrict__ feats,
                                  const float* __restrict__ emb,
                                  _Float16* __restrict__ h16) {
    int wave = (blockIdx.x * blockDim.x + threadIdx.x) >> 5;   // wave32
    int lane = threadIdx.x & 31;
    if (wave >= N_NODES) return;
    const int* f = feats + (size_t)wave * SEQ_LEN;
    float sx = 0.f, sy = 0.f, sz = 0.f, sw = 0.f;
    float mx = -3.0e38f, my = -3.0e38f, mz = -3.0e38f, mw = -3.0e38f;
    int cnt = 0;
    for (int t = 0; t < SEQ_LEN; ++t) {
        int tok = f[t];
        if (tok != 0) {
            ++cnt;
            float4 v = *(const float4*)(emb + (size_t)tok * EMB_DIM + lane * 4);
            sx += v.x; sy += v.y; sz += v.z; sw += v.w;
            mx = fmaxf(mx, v.x); my = fmaxf(my, v.y);
            mz = fmaxf(mz, v.z); mw = fmaxf(mw, v.w);
        } else { // padding row is a zero vector and participates in the max
            mx = fmaxf(mx, 0.f); my = fmaxf(my, 0.f);
            mz = fmaxf(mz, 0.f); mw = fmaxf(mw, 0.f);
        }
    }
    float inv = 1.0f / (float)(cnt > 0 ? cnt : 1);
    _Float16* o = h16 + (size_t)wave * IN_FEATS + lane * 4;
    o[0] = (_Float16)(sx * inv); o[1] = (_Float16)(sy * inv);
    o[2] = (_Float16)(sz * inv); o[3] = (_Float16)(sw * inv);
    _Float16* o2 = o + EMB_DIM;
    o2[0] = (_Float16)mx; o2[1] = (_Float16)my;
    o2[2] = (_Float16)mz; o2[3] = (_Float16)mw;
}

// ---------------- weight conversion (f32 [K][N] -> f16 transposed [N][K]) ----------------
__global__ void convert_w1_kernel(const float* __restrict__ W, _Float16* __restrict__ Wt) {
    int idx = blockIdx.x * blockDim.x + threadIdx.x;        // 256*256
    int n = idx >> 8, k = idx & 255;
    Wt[(size_t)n * IN_FEATS + k] = (_Float16)W[(size_t)k * N_HIDDEN + n];
}

__global__ void convert_w2_kernel(const float* __restrict__ W, _Float16* __restrict__ Wt) {
    int idx = blockIdx.x * blockDim.x + threadIdx.x;        // 48*256
    if (idx >= N_CLS_PAD * N_HIDDEN) return;
    int n = idx >> 8, k = idx & 255;
    Wt[(size_t)n * N_HIDDEN + k] =
        (n < N_CLASSES) ? (_Float16)W[(size_t)k * N_CLASSES + n] : (_Float16)0.0f;
}

// ---------------- WMMA GEMM: C[M,N] = A[M,256] * Bt[N,256]^T ----------------
// one wave per 16x16 output tile, K=256 in 8 steps of v_wmma_f32_16x16x32_f16
__global__ void wmma_gemm_kernel(const _Float16* __restrict__ A,
                                 const _Float16* __restrict__ Bt,
                                 float* __restrict__ C,
                                 int nTileN, int totalTiles, int ldc) {
    int waveId = (blockIdx.x * blockDim.x + threadIdx.x) >> 5;
    int lane   = threadIdx.x & 31;
    if (waveId >= totalTiles) return;
    int mt = waveId / nTileN;
    int nt = waveId % nTileN;
    int half = lane >> 4;          // 0: lanes 0-15, 1: lanes 16-31
    int l15  = lane & 15;

    // A fragment (16x32 f16): lanes 0-15 -> M=lane, elems 0-7 = K+0..7, elems 8-15 = K+16..23
    //                          lanes16-31 -> M=lane-16, K offset +8 / +24
    const _Float16* arow = A  + (size_t)(mt * 16 + l15) * IN_FEATS + half * 8;
    // B fragment (32x16 f16): lanes 0-15 -> N=lane, elems 0-7 = K+0..7, elems 8-15 = K+16..23; lanes16-31 -> K offset +8 / +24
    const _Float16* brow = Bt + (size_t)(nt * 16 + l15) * IN_FEATS + half * 8;

    v8f acc = {};
#pragma unroll
    for (int k0 = 0; k0 < IN_FEATS; k0 += 32) {
        v16h a, b;
        *(float4*)&a         = *(const float4*)(arow + k0);
        *(((float4*)&a) + 1) = *(const float4*)(arow + k0 + 16);
        *(float4*)&b         = *(const float4*)(brow + k0);
        *(((float4*)&b) + 1) = *(const float4*)(brow + k0 + 16);
        acc = __builtin_amdgcn_wmma_f32_16x16x32_f16(
            /*neg_a=*/false, a, /*neg_b=*/false, b,
            /*c_mod=*/(short)0, acc, /*reuse_a=*/false, /*reuse_b=*/false);
    }
    // C/D layout: VGPR v holds (M = v + 8*half, N = lane&15)
    int col  = nt * 16 + l15;
    int row0 = mt * 16 + half * 8;
#pragma unroll
    for (int v = 0; v < 8; ++v)
        C[(size_t)(row0 + v) * ldc + col] = acc[v];
}

// ---------------- edge scatter (message passing) ----------------
__global__ void scatter256_kernel(const float* __restrict__ hw,
                                  const int* __restrict__ src,
                                  const int* __restrict__ dst,
                                  const float* __restrict__ norm_s,
                                  float* __restrict__ agg) {
    int e = blockIdx.x;            // one block (256 threads) per edge
    int c = threadIdx.x;
    int s = src[e], d = dst[e];
    float v = hw[(size_t)s * N_HIDDEN + c] * norm_s[s];
    atomicAdd(&agg[(size_t)d * N_HIDDEN + c], v);
}

__global__ void scatter40_kernel(const float* __restrict__ hw,
                                 const int* __restrict__ src,
                                 const int* __restrict__ dst,
                                 const float* __restrict__ norm_s,
                                 float* __restrict__ agg) {
    int e = blockIdx.x;            // one block (64 threads) per edge
    int c = threadIdx.x;
    if (c >= N_CLASSES) return;
    int s = src[e], d = dst[e];
    float v = hw[(size_t)s * N_CLS_PAD + c] * norm_s[s];
    atomicAdd(&agg[(size_t)d * N_CLASSES + c], v);
}

// ---------------- layer-1 epilogue: *norm_d + b1, ReLU, requantize f16 ----------------
__global__ void relu_bias_kernel(const float* __restrict__ agg,
                                 const float* __restrict__ norm_d,
                                 const float* __restrict__ b1,
                                 _Float16* __restrict__ h16) {
    int idx = blockIdx.x * blockDim.x + threadIdx.x;   // N_NODES * 256
    if (idx >= N_NODES * N_HIDDEN) return;
    int i = idx >> 8, c = idx & 255;
    float v = agg[idx] * norm_d[i] + b1[c];
    h16[idx] = (_Float16)(v > 0.f ? v : 0.f);
}

// ---------------- final: out = agg2 * norm_d + b2 ----------------
__global__ void finalize_kernel(const float* __restrict__ agg,
                                const float* __restrict__ norm_d,
                                const float* __restrict__ b2,
                                float* __restrict__ out) {
    int idx = blockIdx.x * blockDim.x + threadIdx.x;   // N_NODES * 40
    if (idx >= N_NODES * N_CLASSES) return;
    int i = idx / N_CLASSES, c = idx % N_CLASSES;
    out[idx] = agg[idx] * norm_d[i] + b2[c];
}

// ---------------- host side ----------------
extern "C" void kernel_launch(void* const* d_in, const int* in_sizes, int n_in,
                              void* d_out, int out_size, void* d_ws, size_t ws_size,
                              hipStream_t stream) {
    const int*   features = (const int*)  d_in[0];
    const int*   src      = (const int*)  d_in[1];
    const int*   dst      = (const int*)  d_in[2];
    const float* emb      = (const float*)d_in[3];
    const float* W1       = (const float*)d_in[4];
    const float* b1       = (const float*)d_in[5];
    const float* W2       = (const float*)d_in[6];
    const float* b2       = (const float*)d_in[7];
    float*       out      = (float*)d_out;

    // ---- workspace carve-out (256B aligned) ----
    size_t off = 0;
    auto alloc = [&](size_t bytes) -> void* {
        off = (off + 255) & ~(size_t)255;
        void* p = (char*)d_ws + off;
        off += bytes;
        return p;
    };
    float*    deg_s = (float*)   alloc((size_t)N_NODES * 4);                 // -> norm_s
    float*    deg_d = (float*)   alloc((size_t)N_NODES * 4);                 // -> norm_d
    _Float16* h16   = (_Float16*)alloc((size_t)N_NODES * IN_FEATS * 2);      // reused for layer-2 input
    _Float16* w1t   = (_Float16*)alloc((size_t)N_HIDDEN * IN_FEATS * 2);
    _Float16* w2t   = (_Float16*)alloc((size_t)N_CLS_PAD * N_HIDDEN * 2);
    float*    hw1   = (float*)   alloc((size_t)N_NODES * N_HIDDEN * 4);      // reused: hw2 + agg2
    float*    agg1  = (float*)   alloc((size_t)N_NODES * N_HIDDEN * 4);
    float*    hw2   = hw1;                                                   // 50000*48 floats
    float*    agg2  = (float*)((char*)hw1 + (((size_t)N_NODES * N_CLS_PAD * 4 + 255) & ~(size_t)255));

    // ---- 1) degrees & norms ----
    hipMemsetAsync(deg_s, 0, (size_t)N_NODES * 4, stream);
    hipMemsetAsync(deg_d, 0, (size_t)N_NODES * 4, stream);
    edge_degree_kernel<<<(N_EDGES + 255) / 256, 256, 0, stream>>>(src, dst, deg_s, deg_d);
    norm_kernel<<<(2 * N_NODES + 255) / 256, 256, 0, stream>>>(deg_s, 2 * N_NODES); // deg_d starts 48 floats after deg_s ends; its last 48 are skipped

    // ---- 2) embedding pool (wave per node) ----
    embed_pool_kernel<<<N_NODES / 8, 256, 0, stream>>>(features, emb, h16);

    // ---- 3) weight conversion ----
    convert_w1_kernel<<<(N_HIDDEN * IN_FEATS) / 256, 256, 0, stream>>>(W1, w1t);
    convert_w2_kernel<<<(N_CLS_PAD * N_HIDDEN) / 256, 256, 0, stream>>>(W2, w2t);

    // ---- 4) GEMM1: hw1 = h @ W1   (3125 x 16 tiles) ----
    {
        int tiles = (N_NODES / 16) * (N_HIDDEN / 16);
        int blocks = (tiles * 32 + 255) / 256;
        wmma_gemm_kernel<<<blocks, 256, 0, stream>>>(h16, w1t, hw1,
                                                     N_HIDDEN / 16, tiles, N_HIDDEN);
    }

    // ---- 5) scatter layer 1 ----
    hipMemsetAsync(agg1, 0, (size_t)N_NODES * N_HIDDEN * 4, stream);
    scatter256_kernel<<<N_EDGES, 256, 0, stream>>>(hw1, src, dst, deg_s, agg1);

    // ---- 6) epilogue 1: h2 = relu(agg1*norm_d + b1) -> f16 (overwrites h16) ----
    relu_bias_kernel<<<(N_NODES * N_HIDDEN + 255) / 256, 256, 0, stream>>>(agg1, deg_d, b1, h16);

    // ---- 7) GEMM2: hw2 = h2 @ W2  (3125 x 3 tiles, N padded to 48) ----
    {
        int tiles = (N_NODES / 16) * (N_CLS_PAD / 16);
        int blocks = (tiles * 32 + 255) / 256;
        wmma_gemm_kernel<<<blocks, 256, 0, stream>>>(h16, w2t, hw2,
                                                     N_CLS_PAD / 16, tiles, N_CLS_PAD);
    }

    // ---- 8) scatter layer 2 ----
    hipMemsetAsync(agg2, 0, (size_t)N_NODES * N_CLASSES * 4, stream);
    scatter40_kernel<<<N_EDGES, 64, 0, stream>>>(hw2, src, dst, deg_s, agg2);

    // ---- 9) final output ----
    finalize_kernel<<<(N_NODES * N_CLASSES + 255) / 256, 256, 0, stream>>>(agg2, deg_d, b2, out);
}
